// RGCN_9929964388947
// MI455X (gfx1250) — matrix operand order repaired and checked by reading, not compiled, not run on hardware
//
#include <hip/hip_runtime.h>

// ---------------------------------------------------------------------------
// CDNA5 (gfx1250) RGCN inference: WMMA bf16 GEMMs + L2-resident f32 atomics
// ---------------------------------------------------------------------------

typedef __attribute__((ext_vector_type(16))) __bf16 v16bf;
typedef __attribute__((ext_vector_type(8)))  float  v8f;

#define WMMA_BF16(A, B, C) \
  __builtin_amdgcn_wmma_f32_16x16x32_bf16(false, (A), false, (B), (short)0, (C), false, false)

// Pack one 16x32 bf16 A-tile slice for this lane.
// Lane l (l&15 = matrix row m, l>>4 = K-subgroup sel).
// Layout: element j holds K = sel*16 + j (contiguous, matches load_b).
static __device__ __forceinline__ v16bf pack_a(const float* __restrict__ row,
                                               int kc, int sel, float scale) {
  const float* p = row + kc * 32 + sel * 16;
  v16bf a;
#pragma unroll
  for (int j = 0; j < 8; ++j) {
    a[j]     = (__bf16)(p[j] * scale);
    a[j + 8] = (__bf16)(p[j + 8] * scale);
  }
  return a;
}

// B = W^T tile (K x 16). Lane l: column n = l&15, K-half sel = l>>4.
// b[j] = W[n][kc*32 + sel*16 + j]  -> contiguous 16 bf16 = one 32B vector load.
static __device__ __forceinline__ v16bf load_b(const __bf16* __restrict__ W,
                                               int n, int kc, int sel) {
  return *(const v16bf*)(W + (size_t)n * 64 + kc * 32 + sel * 16);
}

// --------------------------- weight f32 -> bf16 ----------------------------
__global__ __launch_bounds__(256) void cvt_bf16_kernel(const float* __restrict__ src,
                                                       __bf16* __restrict__ dst, int n) {
  int i = blockIdx.x * blockDim.x + threadIdx.x;
  if (i < n) dst[i] = (__bf16)src[i];
}

// ------------------- Y = X @ W^T + b (optional ReLU) -----------------------
// One wave computes a 16 x (NT*16) output tile. K = 64 -> 2 WMMAs per N-tile.
template <int NT, int LDO, bool RELU>
__global__ __launch_bounds__(256)
void linear_wmma_kernel(const float* __restrict__ X, const __bf16* __restrict__ W,
                        const float* __restrict__ bias, float* __restrict__ Y,
                        int nRows) {
  const int wave    = threadIdx.x >> 5;
  const int lane    = threadIdx.x & 31;
  const int rowBase = (blockIdx.x * 8 + wave) * 16;
  if (rowBase >= nRows) return;  // wave-uniform exit; EXEC all-ones inside
  const int n   = lane & 15;
  const int sel = lane >> 4;

  const float* Xrow = X + (size_t)(rowBase + n) * 64;
  const v16bf a0 = pack_a(Xrow, 0, sel, 1.0f);
  const v16bf a1 = pack_a(Xrow, 1, sel, 1.0f);

#pragma unroll
  for (int t = 0; t < NT; ++t) {
    v8f c = {};
    c = WMMA_BF16(a0, load_b(W, t * 16 + n, 0, sel), c);
    c = WMMA_BF16(a1, load_b(W, t * 16 + n, 1, sel), c);
    const float bv = bias[t * 16 + n];
#pragma unroll
    for (int r = 0; r < 8; ++r) {   // D: VGPR r -> row (r + sel*8), col n
      const int g = rowBase + r + sel * 8;
      float y = c[r] + bv;
      if (RELU) y = fmaxf(y, 0.0f);
      Y[(size_t)g * LDO + t * 16 + n] = y;
    }
  }
}

// ---- Y = relu( (Acc/deg) @ Wrel^T + brel + H @ Wroot^T + H ) --------------
__global__ __launch_bounds__(256)
void conv_wmma_kernel(const float* __restrict__ Acc, const float* __restrict__ Deg,
                      const float* __restrict__ Hdst,
                      const __bf16* __restrict__ Wrel, const float* __restrict__ Brel,
                      const __bf16* __restrict__ Wroot,
                      float* __restrict__ Y, int nRows) {
  const int wave    = threadIdx.x >> 5;
  const int lane    = threadIdx.x & 31;
  const int rowBase = (blockIdx.x * 8 + wave) * 16;
  if (rowBase >= nRows) return;
  const int n   = lane & 15;
  const int sel = lane >> 4;

  // mean = acc * rcp(max(deg,1)); v_rcp_f32 (~1 ulp) is plenty: result feeds
  // a bf16 (8-bit mantissa) WMMA operand. Avoids the IEEE div sequence.
  const float scale = __builtin_amdgcn_rcpf(fmaxf(Deg[rowBase + n], 1.0f));
  const float* Arow = Acc  + (size_t)(rowBase + n) * 64;
  const float* Hrow = Hdst + (size_t)(rowBase + n) * 64;
  const v16bf a0 = pack_a(Arow, 0, sel, scale);
  const v16bf a1 = pack_a(Arow, 1, sel, scale);
  const v16bf h0 = pack_a(Hrow, 0, sel, 1.0f);
  const v16bf h1 = pack_a(Hrow, 1, sel, 1.0f);

#pragma unroll
  for (int t = 0; t < 4; ++t) {
    v8f c = {};
    c = WMMA_BF16(a0, load_b(Wrel,  t * 16 + n, 0, sel), c);
    c = WMMA_BF16(a1, load_b(Wrel,  t * 16 + n, 1, sel), c);
    c = WMMA_BF16(h0, load_b(Wroot, t * 16 + n, 0, sel), c);
    c = WMMA_BF16(h1, load_b(Wroot, t * 16 + n, 1, sel), c);
    const float bv = Brel[t * 16 + n];
#pragma unroll
    for (int r = 0; r < 8; ++r) {
      const int g = rowBase + r + sel * 8;
      const float y = c[r] + bv + Hdst[(size_t)g * 64 + t * 16 + n];
      Y[(size_t)g * 64 + t * 16 + n] = fmaxf(y, 0.0f);
    }
  }
}

// -------------------- edge scatter: Acc[dst] += H[src] ---------------------
// One wave per edge: 32 lanes x 2 floats = one coalesced 256B row gather and
// 64 f32 atomics that resolve in the 192MB L2 (acc arrays fit).
template <bool DODEG>
__global__ __launch_bounds__(256)
void scatter_kernel(const float* __restrict__ Hsrc, const int* __restrict__ eSrc,
                    const int* __restrict__ eDst, float* __restrict__ Acc,
                    float* __restrict__ Deg, int nEdges) {
  const long long idx = (long long)blockIdx.x * blockDim.x + threadIdx.x;
  if (idx >= (long long)nEdges * 32) return;
  const int e = (int)(idx >> 5);
  const int c = (int)(idx & 31) * 2;
  const int s = eSrc[e];
  const int d = eDst[e];
  const float* h = Hsrc + (size_t)s * 64 + c;
  float* a = Acc + (size_t)d * 64 + c;
  atomicAdd(a,     h[0]);
  atomicAdd(a + 1, h[1]);
  if (DODEG && c == 0) atomicAdd(Deg + d, 1.0f);
}

// ---------------------------------------------------------------------------
extern "C" void kernel_launch(void* const* d_in, const int* in_sizes, int n_in,
                              void* d_out, int out_size, void* d_ws, size_t ws_size,
                              hipStream_t stream) {
  (void)in_sizes; (void)n_in; (void)out_size; (void)ws_size;
  constexpr int NUc = 200000, NIc = 100000, E = 1000000;

  const float* x_user = (const float*)d_in[0];
  const float* x_item = (const float*)d_in[1];
  const int*   ei_u2i = (const int*)d_in[2];   // [2,E]: src=users, dst=items
  const int*   ei_i2u = (const int*)d_in[3];   // [2,E]: src=items, dst=users

  // workspace layout
  char* ws = (char*)d_ws;
  size_t off = 0;
  float* h_u   = (float*)(ws + off); off += (size_t)NUc * 64 * 4;
  float* h_i   = (float*)(ws + off); off += (size_t)NIc * 64 * 4;
  float* x1_u  = (float*)(ws + off); off += (size_t)NUc * 64 * 4;
  float* x1_i  = (float*)(ws + off); off += (size_t)NIc * 64 * 4;
  float* acc_u = (float*)(ws + off); off += (size_t)NUc * 64 * 4;
  float* acc_i = (float*)(ws + off); off += (size_t)NIc * 64 * 4;
  float* deg_u = (float*)(ws + off); off += (size_t)NUc * 4;
  float* deg_i = (float*)(ws + off); off += (size_t)NIc * 4;
  __bf16* wbf  = (__bf16*)(ws + off);          // 12 slots x 4096 bf16
  auto Wslot = [&](int i) { return wbf + (size_t)i * 4096; };

  // ---- convert weights to bf16 (slots: 0 pre_u, 1 pre_i, 2/3 c1u2i rel/root,
  //      4/5 c1i2u, 6/7 c2u2i, 8/9 c2i2u, 10 post_u, 11 post_i)
  cvt_bf16_kernel<<<16, 256, 0, stream>>>((const float*)d_in[4],  Wslot(0), 4096);
  cvt_bf16_kernel<<<16, 256, 0, stream>>>((const float*)d_in[6],  Wslot(1), 4096);
  cvt_bf16_kernel<<<16, 256, 0, stream>>>((const float*)d_in[8],  Wslot(2), 4096);
  cvt_bf16_kernel<<<16, 256, 0, stream>>>((const float*)d_in[10], Wslot(3), 4096);
  cvt_bf16_kernel<<<16, 256, 0, stream>>>((const float*)d_in[11], Wslot(4), 4096);
  cvt_bf16_kernel<<<16, 256, 0, stream>>>((const float*)d_in[13], Wslot(5), 4096);
  cvt_bf16_kernel<<<16, 256, 0, stream>>>((const float*)d_in[14], Wslot(6), 4096);
  cvt_bf16_kernel<<<16, 256, 0, stream>>>((const float*)d_in[16], Wslot(7), 4096);
  cvt_bf16_kernel<<<16, 256, 0, stream>>>((const float*)d_in[17], Wslot(8), 4096);
  cvt_bf16_kernel<<<16, 256, 0, stream>>>((const float*)d_in[19], Wslot(9), 4096);
  cvt_bf16_kernel<<<8,  256, 0, stream>>>((const float*)d_in[20], Wslot(10), 2048);
  cvt_bf16_kernel<<<8,  256, 0, stream>>>((const float*)d_in[22], Wslot(11), 2048);

  const int gridU = (NUc + 127) / 128;   // 8 waves/block * 16 rows/wave
  const int gridI = (NIc + 127) / 128;
  const int gridE = (int)(((long long)E * 32 + 255) / 256);

  // ---- pre linears
  linear_wmma_kernel<4, 64, false><<<gridU, 256, 0, stream>>>(
      x_user, Wslot(0), (const float*)d_in[5], h_u, NUc);
  linear_wmma_kernel<4, 64, false><<<gridI, 256, 0, stream>>>(
      x_item, Wslot(1), (const float*)d_in[7], h_i, NIc);

  // ---- layer 1 scatter (also builds degrees, reused by layer 2)
  hipMemsetAsync(acc_i, 0, (size_t)NIc * 64 * 4, stream);
  hipMemsetAsync(deg_i, 0, (size_t)NIc * 4, stream);
  scatter_kernel<true><<<gridE, 256, 0, stream>>>(h_u, ei_u2i, ei_u2i + E, acc_i, deg_i, E);
  hipMemsetAsync(acc_u, 0, (size_t)NUc * 64 * 4, stream);
  hipMemsetAsync(deg_u, 0, (size_t)NUc * 4, stream);
  scatter_kernel<true><<<gridE, 256, 0, stream>>>(h_i, ei_i2u, ei_i2u + E, acc_u, deg_u, E);

  // ---- layer 1 conv (fused mean + dual GEMM + bias + residual + relu)
  conv_wmma_kernel<<<gridI, 256, 0, stream>>>(acc_i, deg_i, h_i, Wslot(2),
                                              (const float*)d_in[9],  Wslot(3), x1_i, NIc);
  conv_wmma_kernel<<<gridU, 256, 0, stream>>>(acc_u, deg_u, h_u, Wslot(4),
                                              (const float*)d_in[12], Wslot(5), x1_u, NUc);

  // ---- layer 2 scatter (degrees unchanged; same edge sets)
  hipMemsetAsync(acc_i, 0, (size_t)NIc * 64 * 4, stream);
  scatter_kernel<false><<<gridE, 256, 0, stream>>>(x1_u, ei_u2i, ei_u2i + E, acc_i, deg_i, E);
  hipMemsetAsync(acc_u, 0, (size_t)NUc * 64 * 4, stream);
  scatter_kernel<false><<<gridE, 256, 0, stream>>>(x1_i, ei_i2u, ei_i2u + E, acc_u, deg_u, E);

  // ---- layer 2 conv; x2 reuses h buffers (h no longer needed)
  float* x2_i = h_i;
  float* x2_u = h_u;
  conv_wmma_kernel<<<gridI, 256, 0, stream>>>(acc_i, deg_i, x1_i, Wslot(6),
                                              (const float*)d_in[15], Wslot(7), x2_i, NIc);
  conv_wmma_kernel<<<gridU, 256, 0, stream>>>(acc_u, deg_u, x1_u, Wslot(8),
                                              (const float*)d_in[18], Wslot(9), x2_u, NUc);

  // ---- post linears -> d_out (out_u then out_i, flat)
  float* out = (float*)d_out;
  linear_wmma_kernel<2, 32, false><<<gridU, 256, 0, stream>>>(
      x2_u, Wslot(10), (const float*)d_in[21], out, NUc);
  linear_wmma_kernel<2, 32, false><<<gridI, 256, 0, stream>>>(
      x2_i, Wslot(11), (const float*)d_in[23], out + (size_t)NUc * 32, NIc);
}
